// ContrastDrivenFeatureAggregation_3298534883733
// MI455X (gfx1250) — matrix operand order repaired and checked by reading, not compiled, not run on hardware
//
#include <hip/hip_runtime.h>
#include <hip/hip_bf16.h>

typedef __attribute__((ext_vector_type(16))) _Float16 half16;
typedef __attribute__((ext_vector_type(8)))  _Float16 half8;
typedef __attribute__((ext_vector_type(8)))  float    float8;

#define HH 40
#define WW 40
#define LL 1600
#define CC 256
#define KC 2304           // 256 * 9
#define AA 648            // heads * 9 * 9
#define APAD 704          // 648 padded up to multiple of 64 (N block tile)
#define HEADS 8
#define HDIM 32
#define ATT_SCALE 0.17677669529663689f   // 32^-0.5

// ---------------------------------------------------------------------------
// Prep kernels: f32 -> f16 casts / layout changes
// ---------------------------------------------------------------------------

__global__ void cast_f16(const float* __restrict__ a, _Float16* __restrict__ b, int n) {
    int i = blockIdx.x * 256 + threadIdx.x;
    if (i < n) b[i] = (_Float16)a[i];
}

// a: [In, Out] f32 row-major -> bt: [Nalloc, In] f16 (bt[o][i] = a[i][o]; zero pad rows >= Out)
__global__ void transpose_cast(const float* __restrict__ a, _Float16* __restrict__ bt,
                               int In, int Out, int Nalloc) {
    int i = blockIdx.x * 256 + threadIdx.x;
    if (i < Nalloc * In) {
        int o = i / In, ii = i - o * In;
        bt[i] = (o < Out) ? (_Float16)a[ii * Out + o] : (_Float16)0.0f;
    }
}

// a: [C, L] f32 (NCHW) -> b: [L, C] f16 (HWC)
__global__ void nchw2hwc(const float* __restrict__ a, _Float16* __restrict__ b, int C, int L) {
    int i = blockIdx.x * 256 + threadIdx.x;
    if (i < C * L) {
        int l = i / C, c = i - l * C;
        b[i] = (_Float16)a[c * L + l];
    }
}

// a: [L, C] f32 (HWC) -> b: [C, L] f32 (NCHW)
__global__ void hwc2nchw(const float* __restrict__ a, float* __restrict__ b, int C, int L) {
    int i = blockIdx.x * 256 + threadIdx.x;
    if (i < C * L) {
        int c = i / L, l = i - c * L;
        b[i] = a[l * C + c];
    }
}

// ---------------------------------------------------------------------------
// im2col: HWC f16 [L, C] -> patches f16 [L, C*9], patch k = c*9 + kh*3 + kw
// (matches conv_general_dilated_patches channel order (C, kh, kw) and OIHW flat)
// ---------------------------------------------------------------------------
__global__ __launch_bounds__(256) void im2col(const _Float16* __restrict__ src,
                                              _Float16* __restrict__ dst) {
    int l = blockIdx.x;           // pixel
    int c = threadIdx.x;          // channel
    int y = l / WW, x = l - y * WW;
    _Float16* d = dst + (size_t)l * KC + c * 9;
#pragma unroll
    for (int t = 0; t < 9; t++) {
        int ys = y + t / 3 - 1, xs = x + t % 3 - 1;
        _Float16 v = (_Float16)0.0f;
        if (ys >= 0 && ys < HH && xs >= 0 && xs < WW) v = src[(ys * WW + xs) * CC + c];
        d[t] = v;
    }
}

// ---------------------------------------------------------------------------
// WMMA GEMM: C[M, N] = A[M, K] (f16, row major) * Bt[Nalloc, K] (f16, B transposed)
// Block = 128 threads = 4 waves arranged 2(M) x 2(N); wave tile 32x32 (2x2 WMMA),
// block tile 64x64. 4 WMMAs per wave per K-step on 2 A + 2 B fragments.
// epi: 0 = none, 1 = +shift[col], 2 = relu(v*scale[col] + shift[col])
// ---------------------------------------------------------------------------
__global__ __launch_bounds__(128) void gemm_wmma(
    const _Float16* __restrict__ A, const _Float16* __restrict__ Bt,
    float* __restrict__ Cf32, _Float16* __restrict__ Cf16,
    const float* __restrict__ scale, const float* __restrict__ shift,
    int M, int Nvalid, int Kd, int ldc, int epi)
{
    const int wave = threadIdx.x >> 5;
    const int lane = threadIdx.x & 31;
    const int wm = wave >> 1;                 // 0..1
    const int wn = wave & 1;                  // 0..1
    const int row0 = blockIdx.x * 64 + wm * 32;   // wave covers rows [row0, row0+32)
    const int col0 = blockIdx.y * 64 + wn * 32;   // wave covers cols [col0, col0+32)
    const int r  = lane & 15;
    const int kh = lane >> 4;                 // K-half select

    const _Float16* __restrict__ arow0 = A  + (size_t)(row0 + r)      * Kd;
    const _Float16* __restrict__ arow1 = A  + (size_t)(row0 + 16 + r) * Kd;
    const _Float16* __restrict__ brow0 = Bt + (size_t)(col0 + r)      * Kd;
    const _Float16* __restrict__ brow1 = Bt + (size_t)(col0 + 16 + r) * Kd;

    float8 acc00 = {}, acc01 = {}, acc10 = {}, acc11 = {};

#pragma unroll 2
    for (int kk = 0; kk < Kd; kk += 32) {
        if (kk + 128 < Kd) {   // stream-ahead into cache (global_prefetch_b8)
            __builtin_prefetch(arow0 + kk + 128, 0, 0);
            __builtin_prefetch(brow0 + kk + 128, 0, 0);
        }
        // A fragments (16-bit 16x32 layout): lanes 0-15 K0-7/K16-23, lanes 16-31 K8-15/K24-31
        half8 a0lo = *(const half8*)(arow0 + kk + kh * 8);
        half8 a0hi = *(const half8*)(arow0 + kk + 16 + kh * 8);
        half8 a1lo = *(const half8*)(arow1 + kk + kh * 8);
        half8 a1hi = *(const half8*)(arow1 + kk + 16 + kh * 8);
        half16 af0 = __builtin_shufflevector(a0lo, a0hi, 0,1,2,3,4,5,6,7,8,9,10,11,12,13,14,15);
        half16 af1 = __builtin_shufflevector(a1lo, a1hi, 0,1,2,3,4,5,6,7,8,9,10,11,12,13,14,15);
        // B fragments (32x16): lane = column, lanes 0-15 K0-7/K16-23, lanes 16-31 K8-15/K24-31
        half8 b0lo = *(const half8*)(brow0 + kk + kh * 8);
        half8 b0hi = *(const half8*)(brow0 + kk + 16 + kh * 8);
        half8 b1lo = *(const half8*)(brow1 + kk + kh * 8);
        half8 b1hi = *(const half8*)(brow1 + kk + 16 + kh * 8);
        half16 bf0 = __builtin_shufflevector(b0lo, b0hi, 0,1,2,3,4,5,6,7,8,9,10,11,12,13,14,15);
        half16 bf1 = __builtin_shufflevector(b1lo, b1hi, 0,1,2,3,4,5,6,7,8,9,10,11,12,13,14,15);

        acc00 = __builtin_amdgcn_wmma_f32_16x16x32_f16(false, af0, false, bf0, (short)0, acc00, false, false);
        acc01 = __builtin_amdgcn_wmma_f32_16x16x32_f16(false, af0, false, bf1, (short)0, acc01, false, false);
        acc10 = __builtin_amdgcn_wmma_f32_16x16x32_f16(false, af1, false, bf0, (short)0, acc10, false, false);
        acc11 = __builtin_amdgcn_wmma_f32_16x16x32_f16(false, af1, false, bf1, (short)0, acc11, false, false);
    }

    // Epilogue. C/D layout: lane&15 = column, lanes 16-31 hold rows 8-15.
    float8 accs[2][2] = {{acc00, acc01}, {acc10, acc11}};
#pragma unroll
    for (int tn = 0; tn < 2; tn++) {
        const int col = col0 + tn * 16 + r;
        if (col >= Nvalid) continue;
        float sc = 1.0f, sh = 0.0f;
        if (epi >= 1) sh = shift[col];
        if (epi == 2) sc = scale[col];
#pragma unroll
        for (int tm = 0; tm < 2; tm++) {
            const int rowb = row0 + tm * 16 + kh * 8;
#pragma unroll
            for (int i = 0; i < 8; i++) {
                float v = accs[tm][tn][i];
                if (epi == 2)      v = fmaxf(v * sc + sh, 0.0f);
                else if (epi == 1) v = v + sh;
                const size_t idx = (size_t)(rowb + i) * ldc + col;
                if (Cf32) Cf32[idx] = v;
                if (Cf16) Cf16[idx] = (_Float16)v;
            }
        }
    }
}

// ---------------------------------------------------------------------------
// Attention: one block per pixel l; wave n = head, lane d = head-dim element.
// logits: [L, ldl] f32 (n*81 + p*9 + q). v: HWC f16 [L, 256].
// xw out: [L, 9, 256] f32 (p outer, c = n*32+d inner).
// ---------------------------------------------------------------------------
__global__ __launch_bounds__(256) void attn_apply(
    const float* __restrict__ logits, int ldl,
    const _Float16* __restrict__ vhwc, float* __restrict__ xw)
{
    __shared__ _Float16 vs[9 * CC];
    const int l = blockIdx.x;
    const int tid = threadIdx.x;
    const int y = l / WW, x = l - y * WW;

    // stage the 9 unfolded neighbor vectors (zero-padded) into LDS
#pragma unroll
    for (int q = 0; q < 9; q++) {
        int ys = y + q / 3 - 1, xs = x + q % 3 - 1;
        _Float16 v = (_Float16)0.0f;
        if (ys >= 0 && ys < HH && xs >= 0 && xs < WW) v = vhwc[(ys * WW + xs) * CC + tid];
        vs[q * CC + tid] = v;
    }
    __syncthreads();

    const int n = tid >> 5;                   // head
    const int d = tid & 31;                   // head-dim lane
    const float* __restrict__ lg = logits + (size_t)l * ldl + n * 81;
    const _Float16* __restrict__ vh = vs + n * HDIM + d;

    float out[9];
#pragma unroll
    for (int p = 0; p < 9; p++) {
        float w[9];
        float m = -1e30f;
#pragma unroll
        for (int q = 0; q < 9; q++) { w[q] = lg[p * 9 + q] * ATT_SCALE; m = fmaxf(m, w[q]); }
        float s = 0.0f;
#pragma unroll
        for (int q = 0; q < 9; q++) { w[q] = __expf(w[q] - m); s += w[q]; }
        float accv = 0.0f;
#pragma unroll
        for (int q = 0; q < 9; q++) accv += w[q] * (float)vh[q * CC];
        out[p] = accv / s;
    }
#pragma unroll
    for (int p = 0; p < 9; p++)
        xw[(size_t)l * KC + p * CC + n * HDIM + d] = out[p];
}

// ---------------------------------------------------------------------------
// Fold (overlap-add adjoint of unfold): y[l, c] = sum over valid (i,j) of
// xw[(y0+1-i)*W + (x0+1-j), i*3+j, c]. Output f16 HWC.
// ---------------------------------------------------------------------------
__global__ __launch_bounds__(256) void fold_k(const float* __restrict__ xw,
                                              _Float16* __restrict__ yout) {
    const int l = blockIdx.x;
    const int c = threadIdx.x;
    const int y0 = l / WW, x0 = l - y0 * WW;
    float acc = 0.0f;
#pragma unroll
    for (int i = 0; i < 3; i++) {
#pragma unroll
        for (int j = 0; j < 3; j++) {
            int ys = y0 + 1 - i, xs = x0 + 1 - j;
            if (ys >= 0 && ys < HH && xs >= 0 && xs < WW)
                acc += xw[(size_t)(ys * WW + xs) * KC + (i * 3 + j) * CC + c];
        }
    }
    yout[l * CC + c] = (_Float16)acc;
}

// ---------------------------------------------------------------------------
// Host side
// ---------------------------------------------------------------------------
static inline size_t align_up(size_t v, size_t a) { return (v + a - 1) / a * a; }

extern "C" void kernel_launch(void* const* d_in, const int* in_sizes, int n_in,
                              void* d_out, int out_size, void* d_ws, size_t ws_size,
                              hipStream_t stream) {
    const float* x   = (const float*)d_in[0];
    const float* fg  = (const float*)d_in[1];
    const float* bg  = (const float*)d_in[2];
    const float* Wc1 = (const float*)d_in[3];
    const float* s1  = (const float*)d_in[4];
    const float* b1  = (const float*)d_in[5];
    const float* Wc2 = (const float*)d_in[6];
    const float* s2  = (const float*)d_in[7];
    const float* b2  = (const float*)d_in[8];
    const float* Wv  = (const float*)d_in[9];
    const float* bv  = (const float*)d_in[10];
    const float* Wfg = (const float*)d_in[11];
    const float* bfg = (const float*)d_in[12];
    const float* Wbg = (const float*)d_in[13];
    const float* bbg = (const float*)d_in[14];
    const float* Wp  = (const float*)d_in[15];
    const float* bp  = (const float*)d_in[16];
    const float* Wc3 = (const float*)d_in[17];
    const float* s3  = (const float*)d_in[18];
    const float* b3  = (const float*)d_in[19];
    const float* Wc4 = (const float*)d_in[20];
    const float* s4  = (const float*)d_in[21];
    const float* b4  = (const float*)d_in[22];

    // workspace carve-out (256B aligned)
    char* p = (char*)d_ws;
    auto alloc = [&](size_t bytes) -> void* { void* r = (void*)p; p += align_up(bytes, 256); return r; };

    const int WCN = CC * KC;          // 589824 conv weight elements
    _Float16* wc1t = (_Float16*)alloc(WCN * 2);
    _Float16* wc2t = (_Float16*)alloc(WCN * 2);
    _Float16* wc3t = (_Float16*)alloc(WCN * 2);
    _Float16* wc4t = (_Float16*)alloc(WCN * 2);
    _Float16* wvt  = (_Float16*)alloc(CC * CC * 2);
    _Float16* wpt  = (_Float16*)alloc(CC * CC * 2);
    _Float16* wfgt = (_Float16*)alloc(APAD * CC * 2);
    _Float16* wbgt = (_Float16*)alloc(APAD * CC * 2);
    _Float16* xh   = (_Float16*)alloc(LL * CC * 2);
    _Float16* fgh  = (_Float16*)alloc(LL * CC * 2);
    _Float16* bgh  = (_Float16*)alloc(LL * CC * 2);
    _Float16* h1   = (_Float16*)alloc(LL * CC * 2);
    _Float16* h0   = (_Float16*)alloc(LL * CC * 2);
    _Float16* vb   = (_Float16*)alloc(LL * CC * 2);
    _Float16* yb   = (_Float16*)alloc(LL * CC * 2);
    _Float16* x1   = (_Float16*)alloc(LL * CC * 2);
    _Float16* x2   = (_Float16*)alloc(LL * CC * 2);
    _Float16* h3   = (_Float16*)alloc(LL * CC * 2);
    _Float16* P    = (_Float16*)alloc((size_t)LL * KC * 2);
    float*    lgts = (float*)   alloc((size_t)LL * APAD * 4);
    float*    xw   = (float*)   alloc((size_t)LL * KC * 4);
    float*    outh = (float*)   alloc((size_t)LL * CC * 4);
    (void)ws_size;

    const dim3 blk(256);
    const dim3 gblk(128);                     // GEMM: 4 waves / block
    const dim3 gN256(LL / 64, CC / 64);       // (25, 4)
    const dim3 gN704(LL / 64, APAD / 64);     // (25, 11)
    const int LC = LL * CC;                   // 409600

    // --- weight / input prep ---
    cast_f16<<<(WCN + 255) / 256, blk, 0, stream>>>(Wc1, wc1t, WCN);
    cast_f16<<<(WCN + 255) / 256, blk, 0, stream>>>(Wc2, wc2t, WCN);
    cast_f16<<<(WCN + 255) / 256, blk, 0, stream>>>(Wc3, wc3t, WCN);
    cast_f16<<<(WCN + 255) / 256, blk, 0, stream>>>(Wc4, wc4t, WCN);
    transpose_cast<<<(CC * CC + 255) / 256, blk, 0, stream>>>(Wv, wvt, CC, CC, CC);
    transpose_cast<<<(CC * CC + 255) / 256, blk, 0, stream>>>(Wp, wpt, CC, CC, CC);
    transpose_cast<<<(APAD * CC + 255) / 256, blk, 0, stream>>>(Wfg, wfgt, CC, AA, APAD);
    transpose_cast<<<(APAD * CC + 255) / 256, blk, 0, stream>>>(Wbg, wbgt, CC, AA, APAD);
    nchw2hwc<<<LC / 256, blk, 0, stream>>>(x,  xh,  CC, LL);
    nchw2hwc<<<LC / 256, blk, 0, stream>>>(fg, fgh, CC, LL);
    nchw2hwc<<<LC / 256, blk, 0, stream>>>(bg, bgh, CC, LL);

    // --- input CBR x2 ---
    im2col<<<LL, blk, 0, stream>>>(xh, P);
    gemm_wmma<<<gN256, gblk, 0, stream>>>(P, wc1t, nullptr, h1, s1, b1, LL, CC, KC, CC, 2);
    im2col<<<LL, blk, 0, stream>>>(h1, P);
    gemm_wmma<<<gN256, gblk, 0, stream>>>(P, wc2t, nullptr, h0, s2, b2, LL, CC, KC, CC, 2);

    // --- v = h0 @ Wv + bv ---
    gemm_wmma<<<gN256, gblk, 0, stream>>>(h0, wvt, nullptr, vb, nullptr, bv, LL, CC, CC, CC, 1);

    // --- fg attention pass ---
    gemm_wmma<<<gN704, gblk, 0, stream>>>(fgh, wfgt, lgts, nullptr, nullptr, bfg, LL, AA, CC, APAD, 1);
    attn_apply<<<LL, blk, 0, stream>>>(lgts, APAD, vb, xw);
    fold_k<<<LL, blk, 0, stream>>>(xw, yb);
    gemm_wmma<<<gN256, gblk, 0, stream>>>(yb, wpt, nullptr, x1, nullptr, bp, LL, CC, CC, CC, 1);

    // --- bg attention pass (v = x1) ---
    gemm_wmma<<<gN704, gblk, 0, stream>>>(bgh, wbgt, lgts, nullptr, nullptr, bbg, LL, AA, CC, APAD, 1);
    attn_apply<<<LL, blk, 0, stream>>>(lgts, APAD, x1, xw);
    fold_k<<<LL, blk, 0, stream>>>(xw, yb);
    gemm_wmma<<<gN256, gblk, 0, stream>>>(yb, wpt, nullptr, x2, nullptr, bp, LL, CC, CC, CC, 1);

    // --- output CBR x2 ---
    im2col<<<LL, blk, 0, stream>>>(x2, P);
    gemm_wmma<<<gN256, gblk, 0, stream>>>(P, wc3t, nullptr, h3, s3, b3, LL, CC, KC, CC, 2);
    im2col<<<LL, blk, 0, stream>>>(h3, P);
    gemm_wmma<<<gN256, gblk, 0, stream>>>(P, wc4t, outh, nullptr, s4, b4, LL, CC, KC, CC, 2);

    // --- HWC f32 -> NCHW f32 output ---
    hwc2nchw<<<LC / 256, blk, 0, stream>>>(outh, (float*)d_out, CC, LL);
    (void)in_sizes; (void)n_in; (void)out_size;
}
